// FixedSemanticCodebook_68023692034240
// MI455X (gfx1250) — matrix operand order repaired and checked by reading, not compiled, not run on hardware
//
#include <hip/hip_runtime.h>
#include <hip/hip_bf16.h>
#include <math.h>

// ---------------------------------------------------------------------------
// VQ-VAE vector quantization for MI455X (gfx1250, wave32, WMMA).
//
// Shapes: inputs [16,256,32,32] f32, embeddings [8192,256] f32 (L2-normalized)
//   N = 16384 flat rows, C = 256, K = 8192 codes.
// Core: argmin_k(||e_k||^2 - 2 x.e_k) via split-bf16 WMMA (3x bf16 GEMM with
// f32 accumulate ~ fp32 accuracy), argmin fused into GEMM epilogue.
// Codebook stream staged to LDS with CDNA5 async global->LDS copies.
// ---------------------------------------------------------------------------

typedef unsigned int u32;
typedef __attribute__((ext_vector_type(16))) __bf16 v16bf;
typedef __attribute__((ext_vector_type(8)))  float  v8f;
typedef int v4i __attribute__((vector_size(16)));   // matches builtin proto

#define VQ_N      16384
#define VQ_K      8192
#define VQ_C      256
#define VQ_MT     512         // K / 16 codebook tiles
#define VQ_NT     1024        // N / 16 row tiles
#define VQ_QELEMS 4194304     // 16*256*32*32

#if __has_builtin(__builtin_amdgcn_global_load_async_to_lds_b128)
#define VQ_HAS_ASYNC_LDS 1
#else
#define VQ_HAS_ASYNC_LDS 0
#endif

typedef __attribute__((address_space(1))) v4i* as1_v4i_ptr;
typedef __attribute__((address_space(3))) v4i* as3_v4i_ptr;

// ---------------------------- bf16 split helpers ---------------------------

__device__ __forceinline__ u32 f32_to_bf16_bits(float f) {
  u32 u = __builtin_bit_cast(u32, f);
  u32 lsb = (u >> 16) & 1u;
  u += 0x7fffu + lsb;                 // round-to-nearest-even
  return u >> 16;
}
__device__ __forceinline__ float bf16_bits_to_f32(u32 b) {
  return __builtin_bit_cast(float, b << 16);
}
__device__ __forceinline__ void split_bf16(float x, u32& hi, u32& lo) {
  hi = f32_to_bf16_bits(x);
  float r = x - bf16_bits_to_f32(hi);
  lo = f32_to_bf16_bits(r);
}

// ----------------------- async global->LDS staging -------------------------
// LDS generic pointers: low 32 bits are the LDS byte offset (ISA aperture
// rule), so truncate+inttoptr into AS(3). Global generic VA is the AS(1) VA.

__device__ __forceinline__ void async_copy16(const u32* g, u32* l) {
#if VQ_HAS_ASYNC_LDS
  __builtin_amdgcn_global_load_async_to_lds_b128(
      (as1_v4i_ptr)(uintptr_t)g,
      (as3_v4i_ptr)(unsigned)(uintptr_t)l,
      0, 0);
#else
  *(uint4*)l = *(const uint4*)g;      // fallback: load + ds_store
#endif
}

__device__ __forceinline__ void wait_async_all() {
#if VQ_HAS_ASYNC_LDS
#if __has_builtin(__builtin_amdgcn_s_wait_asynccnt)
  __builtin_amdgcn_s_wait_asynccnt(0);
#else
  asm volatile("s_wait_asynccnt 0x0" ::: "memory");
#endif
#endif
}

// --------------------- kernel 0a: pack codebook A-fragments ----------------
// A-layout (16-bit A 16x32, ISA 7.12.2): lane l: row M=l&15, g=l>>4,
// VGPR v holds K pair base = (v<4?0:16) + g*8 + (v&3)*2.
// Linear t = mt*2048 + s*256 + lane*8 + v  (one dword = bf16 pair).
__global__ __launch_bounds__(256) void vq_pack_emb(
    const float* __restrict__ emb, u32* __restrict__ EAh, u32* __restrict__ EAl) {
  const u32 t = blockIdx.x * 256u + threadIdx.x;       // < 1,048,576
  const int v    = t & 7;
  const int lane = (t >> 3) & 31;
  const int s    = (t >> 8) & 7;
  const int mt   = t >> 11;
  const int m  = mt * 16 + (lane & 15);
  const int gg = lane >> 4;
  const int Kk = ((v < 4) ? 0 : 16) + gg * 8 + (v & 3) * 2;
  const int c  = s * 32 + Kk;
  const float e0 = emb[(size_t)m * VQ_C + c];
  const float e1 = emb[(size_t)m * VQ_C + c + 1];
  u32 h0, l0, h1, l1;
  split_bf16(e0, h0, l0);
  split_bf16(e1, h1, l1);
  EAh[t] = (h1 << 16) | h0;
  EAl[t] = (l1 << 16) | l0;
}

// ------------------------ kernel 0b: pack x B-fragments --------------------
// B-layout (16-bit B 32x16): lane l: col N=l&15, g=l>>4, VGPR v holds
// K pair base = (v<4?0:16) + g*8 + (v&3)*2. Includes the NCHW -> [N,C] transpose.
__global__ __launch_bounds__(256) void vq_pack_x(
    const float* __restrict__ in, u32* __restrict__ XBh, u32* __restrict__ XBl) {
  const u32 t = blockIdx.x * 256u + threadIdx.x;       // < 2,097,152
  const int v    = t & 7;
  const int lane = (t >> 3) & 31;
  const int s    = (t >> 8) & 7;
  const int nt   = t >> 11;
  const int n  = nt * 16 + (lane & 15);
  const int gg = lane >> 4;
  const int Kk = ((v < 4) ? 0 : 16) + gg * 8 + (v & 3) * 2;
  const int c  = s * 32 + Kk;
  const int bI = n >> 10, h = (n >> 5) & 31, w = n & 31;
  const size_t off = (size_t)bI * 262144 + (size_t)c * 1024 + (size_t)h * 32 + w;
  const float x0 = in[off];
  const float x1 = in[off + 1024];                     // c+1
  u32 h0, l0, h1, l1;
  split_bf16(x0, h0, l0);
  split_bf16(x1, h1, l1);
  XBh[t] = (h1 << 16) | h0;
  XBl[t] = (l1 << 16) | l0;
}

// ------------------------ kernel 0c: ||e_k||^2 -----------------------------
__global__ __launch_bounds__(256) void vq_enorm(
    const float* __restrict__ emb, float* __restrict__ enorm2) {
  __shared__ float red[256];
  const int k = blockIdx.x, c = threadIdx.x;
  const float v = emb[(size_t)k * VQ_C + c];
  red[c] = v * v;
  __syncthreads();
  for (int st = 128; st > 0; st >>= 1) {
    if (c < st) red[c] += red[c + st];
    __syncthreads();
  }
  if (c == 0) enorm2[k] = red[0];
}

// --------------------- kernel 1: WMMA GEMM + fused argmin ------------------
// Block = 256 threads = 8 waves; wave w handles x row-tile nt = blockIdx*8+w,
// keeps its B fragments resident in VGPRs, and loops over all 512 codebook
// tiles staged into double-buffered LDS with async global->LDS copies.
__global__ __launch_bounds__(256) void vq_argmax(
    const u32* __restrict__ EAh, const u32* __restrict__ EAl,
    const u32* __restrict__ XBh, const u32* __restrict__ XBl,
    const float* __restrict__ enorm2,
    int* __restrict__ idx_out, float* __restrict__ idxf_out) {
  __shared__ u32 ldsH[2][2048];   // 8 slabs * 32 lanes * 8 dwords
  __shared__ u32 ldsL[2][2048];

  const int tid  = threadIdx.x;
  const int lane = tid & 31;
  const int wv   = tid >> 5;
  const int nt   = blockIdx.x * 8 + wv;
  const int g    = lane >> 4;

  // Resident x fragments (hi + lo): 128 VGPRs.
  v16bf xh[8], xl[8];
  {
    const size_t base = (size_t)nt * 2048 + (size_t)lane * 8;
#pragma unroll
    for (int s = 0; s < 8; ++s) {
      xh[s] = *reinterpret_cast<const v16bf*>(XBh + base + s * 256);
      xl[s] = *reinterpret_cast<const v16bf*>(XBl + base + s * 256);
    }
  }

  float best[8];
  int   bidx[8];
#pragma unroll
  for (int r = 0; r < 8; ++r) { best[r] = 3.0e38f; bidx[r] = 0; }

  // Prologue: stage codebook tile 0 into LDS buffer 0.
  {
    const size_t goff = (size_t)tid * 8;
    async_copy16(EAh + goff,     &ldsH[0][tid * 8]);
    async_copy16(EAh + goff + 4, &ldsH[0][tid * 8 + 4]);
    async_copy16(EAl + goff,     &ldsL[0][tid * 8]);
    async_copy16(EAl + goff + 4, &ldsL[0][tid * 8 + 4]);
  }
  wait_async_all();
  __syncthreads();

  for (int mt = 0; mt < VQ_MT; ++mt) {
    const int buf = mt & 1;
    // Kick off async staging of the next tile into the other buffer;
    // it runs behind the 24 WMMAs below.
    if (mt + 1 < VQ_MT) {
      const size_t goff = (size_t)(mt + 1) * 2048 + (size_t)tid * 8;
      const int nb = buf ^ 1;
      async_copy16(EAh + goff,     &ldsH[nb][tid * 8]);
      async_copy16(EAh + goff + 4, &ldsH[nb][tid * 8 + 4]);
      async_copy16(EAl + goff,     &ldsL[nb][tid * 8]);
      async_copy16(EAl + goff + 4, &ldsL[nb][tid * 8 + 4]);
    }

    // D = E_tile (16xC slab, A operand) x X_tile^T (B operand).
    // Split-bf16: dot = eh*xh + eh*xl + el*xh  (3 independent acc chains).
    v8f a0 = {}, a1 = {}, a2 = {};
#pragma unroll
    for (int s = 0; s < 8; ++s) {
      const v16bf eh = *reinterpret_cast<const v16bf*>(&ldsH[buf][(s * 32 + lane) * 8]);
      const v16bf el = *reinterpret_cast<const v16bf*>(&ldsL[buf][(s * 32 + lane) * 8]);
      a0 = __builtin_amdgcn_wmma_f32_16x16x32_bf16(false, eh, false, xh[s],
                                                   (short)0, a0, false, false);
      a1 = __builtin_amdgcn_wmma_f32_16x16x32_bf16(false, eh, false, xl[s],
                                                   (short)0, a1, false, false);
      a2 = __builtin_amdgcn_wmma_f32_16x16x32_bf16(false, el, false, xh[s],
                                                   (short)0, a2, false, false);
    }

    // D layout: lane owns x-row (lane&15); slot r is codebook row mt*16+g*8+r.
    const v8f en = *reinterpret_cast<const v8f*>(enorm2 + mt * 16 + g * 8);
#pragma unroll
    for (int r = 0; r < 8; ++r) {
      const float d = en[r] - 2.0f * (a0[r] + a1[r] + a2[r]);
      const int   m = mt * 16 + g * 8 + r;
      if (d < best[r]) { best[r] = d; bidx[r] = m; }
    }

    // Next tile fully in LDS + everyone done reading this one.
    wait_async_all();
    __syncthreads();
  }

  // Reduce 8 slots in-lane (lowest index wins ties).
  float bv = best[0];
  int   bi = bidx[0];
#pragma unroll
  for (int r = 1; r < 8; ++r) {
    if (best[r] < bv || (best[r] == bv && bidx[r] < bi)) { bv = best[r]; bi = bidx[r]; }
  }
  // Combine lane halves (l <-> l^16) via ds_swizzle SWAPX16 (0x401F).
  {
    const int ovb = __builtin_amdgcn_ds_swizzle(__builtin_bit_cast(int, bv), 0x401f);
    const int oib = __builtin_amdgcn_ds_swizzle(bi, 0x401f);
    const float ov = __builtin_bit_cast(float, ovb);
    if (ov < bv || (ov == bv && oib < bi)) { bv = ov; bi = oib; }
  }
  if (lane < 16) {
    const int n = nt * 16 + lane;
    idx_out[n]  = bi;
    idxf_out[n] = (float)bi;
  }
}

// ----------- kernel 2: gather quantized, NHWC->NCHW write, loss, hist ------
__global__ __launch_bounds__(256) void vq_quant_loss(
    const float* __restrict__ in, const float* __restrict__ emb,
    const int* __restrict__ idx, float* __restrict__ out,
    float* __restrict__ loss_acc, int* __restrict__ counts) {
  __shared__ float red[256];
  const int n = blockIdx.x;
  const int c = threadIdx.x;
  const int k = idx[n];
  const int bI = n >> 10, h = (n >> 5) & 31, w = n & 31;
  const size_t off = (size_t)bI * 262144 + (size_t)c * 1024 + (size_t)h * 32 + w;
  const float e = emb[(size_t)k * VQ_C + c];
  const float x = in[off];
  out[off] = e;                       // quantized_ste == quantized numerically
  const float d = e - x;
  red[c] = d * d;
  __syncthreads();
  for (int st = 128; st > 0; st >>= 1) {
    if (c < st) red[c] += red[c + st];
    __syncthreads();
  }
  if (c == 0) {
    atomicAdd(loss_acc, red[0]);
    atomicAdd(&counts[k], 1);
  }
}

// --------------------- kernel 3: loss scale + perplexity -------------------
__global__ __launch_bounds__(256) void vq_finalize(
    const int* __restrict__ counts, const float* __restrict__ loss_acc,
    float* __restrict__ out_scalars) {
  __shared__ float red[256];
  float s = 0.0f;
  for (int k = threadIdx.x; k < VQ_K; k += 256) {
    const float p = (float)counts[k] * (1.0f / (float)VQ_N);
    s += p * logf(p + 1e-10f);
  }
  red[threadIdx.x] = s;
  __syncthreads();
  for (int st = 128; st > 0; st >>= 1) {
    if (threadIdx.x < st) red[threadIdx.x] += red[threadIdx.x + st];
    __syncthreads();
  }
  if (threadIdx.x == 0) {
    // loss = (0.25 + 1.0) * mean((q-x)^2) over N*C elements
    out_scalars[0] = 1.25f * loss_acc[0] / (float)(VQ_N * VQ_C);
    out_scalars[1] = expf(-red[0]);
  }
}

// ------------------------------- launcher ----------------------------------
extern "C" void kernel_launch(void* const* d_in, const int* in_sizes, int n_in,
                              void* d_out, int out_size, void* d_ws, size_t ws_size,
                              hipStream_t stream) {
  const float* in  = (const float*)d_in[0];   // [16,256,32,32]
  const float* emb = (const float*)d_in[1];   // [8192,256]
  float* out = (float*)d_out;
  char*  ws  = (char*)d_ws;

  // Workspace layout (all offsets 256B-aligned):
  u32*   EAh      = (u32*)(ws + 0);                         // 4 MiB
  u32*   EAl      = (u32*)(ws + (4u << 20));                // 4 MiB
  u32*   XBh      = (u32*)(ws + (8u << 20));                // 8 MiB
  u32*   XBl      = (u32*)(ws + (16u << 20));               // 8 MiB
  float* enorm2   = (float*)(ws + (24u << 20));             // 32 KiB
  int*   idx      = (int*)(ws + (24u << 20) + (32u << 10)); // 64 KiB
  int*   counts   = (int*)(ws + (24u << 20) + (96u << 10)); // 32 KiB
  float* loss_acc = (float*)(ws + (24u << 20) + (128u << 10));

  // Zero the histogram + loss accumulator (ws is poisoned, not re-zeroed).
  (void)hipMemsetAsync(counts, 0, (32u << 10) + 256, stream);

  vq_pack_emb<<<4096, 256, 0, stream>>>(emb, EAh, EAl);
  vq_pack_x  <<<8192, 256, 0, stream>>>(in, XBh, XBl);
  vq_enorm   <<<VQ_K, 256, 0, stream>>>(emb, enorm2);

  // d_out layout: quantized [VQ_QELEMS] | loss | perplexity | indices [N]
  vq_argmax<<<VQ_NT / 8, 256, 0, stream>>>(EAh, EAl, XBh, XBl, enorm2,
                                           idx, out + VQ_QELEMS + 2);
  vq_quant_loss<<<VQ_N, 256, 0, stream>>>(in, emb, idx, out, loss_acc, counts);
  vq_finalize<<<1, 256, 0, stream>>>(counts, loss_acc, out + VQ_QELEMS);
}
